// BiAttnModel_64647847739475
// MI455X (gfx1250) — matrix operand order repaired and checked by reading, not compiled, not run on hardware
//
#include <hip/hip_runtime.h>
#include <hip/hip_bf16.h>

typedef __attribute__((ext_vector_type(16))) _Float16 v16h;
typedef __attribute__((ext_vector_type(8)))  _Float16 v8h;
typedef __attribute__((ext_vector_type(8)))  float    v8f;

#define UU    2048
#define Dm    32
#define Bsz   16
#define NPAIR 3
#define NE    (Bsz*UU*Dm)        /* 1,048,576 elems per modality */
#define PB    (NPAIR*Bsz)        /* 48 (pair,batch) combos       */
#define SUMN  (PB*UU)            /* 98,304 per-(p,b)-per-j array */

// ---------------------------------------------------------------------------
// K0: f32 -> f16 conversion of the three embedding tensors (read once, used 48x)
// ---------------------------------------------------------------------------
__global__ __launch_bounds__(256) void cvt_kernel(const float* __restrict__ a,
                                                  const float* __restrict__ v,
                                                  const float* __restrict__ l,
                                                  _Float16* __restrict__ ha,
                                                  _Float16* __restrict__ hv,
                                                  _Float16* __restrict__ hl) {
  for (int i = blockIdx.x * blockDim.x + threadIdx.x; i < NE;
       i += gridDim.x * blockDim.x) {
    ha[i] = (_Float16)a[i];
    hv[i] = (_Float16)v[i];
    hl[i] = (_Float16)l[i];
  }
}

__global__ __launch_bounds__(256) void zero_kernel(float* __restrict__ p, int n) {
  int i = blockIdx.x * blockDim.x + threadIdx.x;
  if (i < n) p[i] = 0.f;
}

// ---------------------------------------------------------------------------
// K1: per (pair, batch): S = f1 * f2^T via v_wmma_f32_16x16x32_f16.
// Accumulate Z_j = row sums of exp(S) and C_j = column sums of exp(S).
// Branch-free inner loop, 2x unrolled so the exp/accumulate of tile 0
// overlaps the WMMA of tile 1 (hides the WMMA->VALU hazard NOPs).
// Grid: (16 row-tiles, 16 batches, 3 pairs); wave w owns rows [i0+16w, i0+16w+16).
// ---------------------------------------------------------------------------
__global__ __launch_bounds__(256) void score_kernel(const _Float16* __restrict__ ha,
                                                    const _Float16* __restrict__ hv,
                                                    const _Float16* __restrict__ hl,
                                                    float* __restrict__ Z,
                                                    float* __restrict__ Cc) {
  __shared__ float colAcc[UU];                       // 8 KB
  const int p  = blockIdx.z;
  const int b  = blockIdx.y;
  const int i0 = blockIdx.x * 128;
  const _Float16* f1 = ((p == 2) ? hv : ha) + (size_t)b * (UU * Dm);
  const _Float16* f2 = ((p == 0) ? hv : hl) + (size_t)b * (UU * Dm);
  const int tid  = threadIdx.x;
  const int wave = tid >> 5;
  const int lane = tid & 31;
  const int m    = lane & 15;                        // row/col within tile
  const int hf   = lane >> 4;                        // half-wave
  const int base = (p * Bsz + b) * UU;

  for (int j = tid; j < UU; j += 256) colAcc[j] = 0.f;
  __syncthreads();

  // A fragment for rows i0 + wave*16 + m, built once.
  // A layout (contiguous, matches B): half0 lanes: K 0..15; half1: K 16..31.
  const _Float16* arow = f1 + (size_t)(i0 + wave * 16 + m) * Dm;
  v16h afrag;
  {
    v8h lo = *(const v8h*)(arow + hf * 16);
    v8h hi = *(const v8h*)(arow + hf * 16 + 8);
#pragma unroll
    for (int e = 0; e < 8; ++e) { afrag[e] = lo[e]; afrag[e + 8] = hi[e]; }
  }

  v8f rowAcc = {0.f, 0.f, 0.f, 0.f, 0.f, 0.f, 0.f, 0.f};
  const _Float16* bbase = f2 + (size_t)m * Dm + hf * 16;  // column n=m of f2^T

  for (int j0 = 0; j0 < UU; j0 += 32) {
    v16h bfrag0 = *(const v16h*)(bbase + (size_t)j0 * Dm);
    v16h bfrag1 = *(const v16h*)(bbase + (size_t)(j0 + 16) * Dm);

    v8f c0 = {0.f, 0.f, 0.f, 0.f, 0.f, 0.f, 0.f, 0.f};
    v8f c1 = {0.f, 0.f, 0.f, 0.f, 0.f, 0.f, 0.f, 0.f};
    c0 = __builtin_amdgcn_wmma_f32_16x16x32_f16(false, afrag, false, bfrag0,
                                                (short)0, c0, false, false);
    c1 = __builtin_amdgcn_wmma_f32_16x16x32_f16(false, afrag, false, bfrag1,
                                                (short)0, c1, false, false);
    // c[r] at lane: D[r + 8*hf][m]
    float colp0 = 0.f, colp1 = 0.f;
#pragma unroll
    for (int r = 0; r < 8; ++r) {
      float e = __expf(c0[r]);
      rowAcc[r] += e;
      colp0 += e;
    }
    atomicAdd(&colAcc[j0 + m], colp0);               // ds_add_f32
#pragma unroll
    for (int r = 0; r < 8; ++r) {
      float e = __expf(c1[r]);
      rowAcc[r] += e;
      colp1 += e;
    }
    atomicAdd(&colAcc[j0 + 16 + m], colp1);
  }

  // Full row sums: each D-row lives across the 16 lanes of one half-wave.
#pragma unroll
  for (int mask = 1; mask < 16; mask <<= 1) {
#pragma unroll
    for (int r = 0; r < 8; ++r)
      rowAcc[r] += __shfl_xor(rowAcc[r], mask, 32);
  }
  if (m == 0) {
#pragma unroll
    for (int r = 0; r < 8; ++r)
      Z[base + i0 + wave * 16 + hf * 8 + r] = rowAcc[r];  // exclusive owner
  }

  __syncthreads();
  for (int j = tid; j < UU; j += 256)
    atomicAdd(&Cc[base + j], colAcc[j]);             // global_atomic_add_f32
}

// ---------------------------------------------------------------------------
// K2: recompute row-0 / col-0 exps in f32 (32-MAC dots, trivial), then
//     O1[d] = sum_j (exp(S[0,j])/C_j) f2[j,d]; O2[d] = sum_j (exp(S[j,0])/Z_j) f1[j,d]
//     Bi[b, pair, :] = [O1*f1[0,:], O2*f2[0,:]]
// ---------------------------------------------------------------------------
__global__ __launch_bounds__(256) void ovec_kernel(const float* __restrict__ a,
                                                   const float* __restrict__ v,
                                                   const float* __restrict__ l,
                                                   const float* __restrict__ Z,
                                                   const float* __restrict__ Cc,
                                                   float* __restrict__ Bi) {
  const int p = blockIdx.x;
  const int b = blockIdx.y;
  const float* f1 = ((p == 2) ? v : a) + (size_t)b * (UU * Dm);
  const float* f2 = ((p == 0) ? v : l) + (size_t)b * (UU * Dm);
  const int base = (p * Bsz + b) * UU;
  const int t = threadIdx.x;

  __shared__ float w1[UU];                           // exp(S[0,j]) / C_j
  __shared__ float w2[UU];                           // exp(S[j,0]) / Z_j
  __shared__ float f10[Dm], f20[Dm];
  if (t < Dm) { f10[t] = f1[t]; f20[t] = f2[t]; }
  __syncthreads();

  for (int j = t; j < UU; j += 256) {
    float d1 = 0.f, d2 = 0.f;
#pragma unroll
    for (int k = 0; k < Dm; ++k) {
      d1 += f10[k] * f2[j * Dm + k];                 // S[0, j]
      d2 += f20[k] * f1[j * Dm + k];                 // S[j, 0]
    }
    w1[j] = __expf(d1) / Cc[base + j];
    w2[j] = __expf(d2) / Z[base + j];
  }
  __syncthreads();

  const int d = t & 31;
  const int g = t >> 5;                              // 8 j-groups
  float acc1 = 0.f, acc2 = 0.f;
  for (int j = g; j < UU; j += 8) {
    acc1 += w1[j] * f2[j * Dm + d];
    acc2 += w2[j] * f1[j * Dm + d];
  }
  __shared__ float s1[256], s2[256];
  s1[t] = acc1; s2[t] = acc2;
  __syncthreads();
  if (g == 0) {
    for (int gg = 1; gg < 8; ++gg) { acc1 += s1[gg * 32 + d]; acc2 += s2[gg * 32 + d]; }
    float* out = Bi + (b * NPAIR + p) * 64;
    out[d]      = acc1 * f1[d];                      // A1 = O1 * f1[0,:]
    out[32 + d] = acc2 * f2[d];                      // A2 = O2 * f2[0,:]
  }
}

// ---------------------------------------------------------------------------
// K3: Ci = tanh(Bi @ fc1_w^T + fc1_b) @ fc2_w^T; alpha = softmax over batch;
//     CCA[b, :] = sum_k alpha[b,k] * Bi[b,k,:]
// ---------------------------------------------------------------------------
__global__ __launch_bounds__(256) void head_kernel(const float* __restrict__ Bi,
                                                   const float* __restrict__ fc1w,
                                                   const float* __restrict__ fc1b,
                                                   const float* __restrict__ fc2w,
                                                   float* __restrict__ CCA) {
  __shared__ float Ci[PB];
  __shared__ float alpha[PB];
  const int t = threadIdx.x;
  if (t < PB) {                                      // t = b*3 + k
    const float* x = Bi + t * 64;
    float ci = 0.f;
    for (int o = 0; o < 64; ++o) {
      float h = fc1b[o];
      for (int d2 = 0; d2 < 64; ++d2) h += x[d2] * fc1w[o * 64 + d2];
      ci += tanhf(h) * fc2w[o];
    }
    Ci[t] = ci;
  }
  __syncthreads();
  if (t < NPAIR) {                                   // softmax over the 16 batches, per k
    float mx = -1e30f;
    for (int bb = 0; bb < Bsz; ++bb) mx = fmaxf(mx, Ci[bb * NPAIR + t]);
    float s = 0.f;
    for (int bb = 0; bb < Bsz; ++bb) {
      float e = __expf(Ci[bb * NPAIR + t] - mx);
      alpha[bb * NPAIR + t] = e; s += e;
    }
    float inv = 1.f / s;
    for (int bb = 0; bb < Bsz; ++bb) alpha[bb * NPAIR + t] *= inv;
  }
  __syncthreads();
  for (int idx = t; idx < Bsz * 64; idx += 256) {
    int bb = idx >> 6, d = idx & 63;
    float sum = 0.f;
    for (int k = 0; k < NPAIR; ++k)
      sum += alpha[bb * NPAIR + k] * Bi[(bb * NPAIR + k) * 64 + d];
    CCA[idx] = sum;
  }
}

// ---------------------------------------------------------------------------
// K4: out[b, u, :] = CCA[b, :] broadcast over u (float4 stores, ~8 MB)
// ---------------------------------------------------------------------------
__global__ __launch_bounds__(256) void bcast_kernel(const float* __restrict__ CCA,
                                                    float4* __restrict__ out) {
  const int i  = blockIdx.x * 256 + threadIdx.x;     // 524,288 float4s
  const int b  = i >> 15;                            // 2048*16 float4 per batch
  const int d4 = i & 15;                             // 16 float4 per 64-vec
  const float4* c4 = (const float4*)CCA;
  out[i] = c4[b * 16 + d4];
}

// ---------------------------------------------------------------------------
extern "C" void kernel_launch(void* const* d_in, const int* in_sizes, int n_in,
                              void* d_out, int out_size, void* d_ws, size_t ws_size,
                              hipStream_t stream) {
  const float* a    = (const float*)d_in[0];
  const float* v    = (const float*)d_in[1];
  const float* l    = (const float*)d_in[2];
  const float* fc1w = (const float*)d_in[3];
  const float* fc1b = (const float*)d_in[4];
  const float* fc2w = (const float*)d_in[5];

  // Workspace carve-up (~6.8 MB total)
  _Float16* ha = (_Float16*)d_ws;
  _Float16* hv = ha + NE;
  _Float16* hl = hv + NE;
  float* Z   = (float*)(hl + NE);                    // row sums of exp(S)
  float* Cc  = Z + SUMN;                             // col sums of exp(S) (atomic)
  float* Bi  = Cc + SUMN;                            // [16][3][64]
  float* CCA = Bi + PB * 64;                         // [16][64]

  cvt_kernel <<<1024, 256, 0, stream>>>(a, v, l, ha, hv, hl);
  zero_kernel<<<(SUMN + 255) / 256, 256, 0, stream>>>(Cc, SUMN);
  score_kernel<<<dim3(16, Bsz, NPAIR), 256, 0, stream>>>(ha, hv, hl, Z, Cc);
  ovec_kernel<<<dim3(NPAIR, Bsz), 256, 0, stream>>>(a, v, l, Z, Cc, Bi);
  head_kernel<<<1, 256, 0, stream>>>(Bi, fc1w, fc1b, fc2w, CCA);
  bcast_kernel<<<2048, 256, 0, stream>>>(CCA, (float4*)d_out);
}
